// ModelWithCRFLoss_52862457479797
// MI455X (gfx1250) — compile-verified, matrix operand order checked by reading, not run on hardware
//
#include <hip/hip_runtime.h>

typedef _Float16 v16h __attribute__((ext_vector_type(16)));
typedef float    v8f  __attribute__((ext_vector_type(8)));
typedef float    v4f  __attribute__((ext_vector_type(4)));

#define Bsz 128
#define Tsz 1024
#define Nst 64

// -------------------------------------------------------------------------
// Kernel 1: per-batch sequence score (unary gather + transition gather)
// -------------------------------------------------------------------------
__global__ void crf_seq_score_kernel(const float* __restrict__ pot,
                                     const int* __restrict__ tags,
                                     const int* __restrict__ seqlen,
                                     const float* __restrict__ K,
                                     float* __restrict__ seq_out) {
    const int b   = blockIdx.x;
    const int tid = threadIdx.x;
    const int L   = seqlen[b];
    const int*   tg = tags + (size_t)b * Tsz;
    const float* pb = pot  + (size_t)b * Tsz * Nst;
    float acc = 0.0f;
    for (int t = tid; t < Tsz; t += 256) {
        if (t < L) {
            int tag = tg[t];
            acc += pb[(size_t)t * Nst + tag];
            if (t >= 1) acc += K[tg[t - 1] * Nst + tag];
        }
    }
    __shared__ float red[256];
    red[tid] = acc;
    __syncthreads();
    #pragma unroll
    for (int s = 128; s > 0; s >>= 1) {
        if (tid < s) red[tid] += red[tid + s];
        __syncthreads();
    }
    if (tid == 0) seq_out[b] = red[0];
}

// -------------------------------------------------------------------------
// Kernel 2: forward-algorithm scan in exp space via WMMA.
//   One wave handles 16 batches (batch = N / lanes).
//   A = exp(K)^T : constant fragments. B rebuilt per step by a pure
//   per-lane repack of D. Pot loads software-pipelined one step ahead.
//   Rescale uses exact power-of-two scaling (no div, no log in the loop).
// -------------------------------------------------------------------------
__global__ __launch_bounds__(32)
void crf_forward_scan_kernel(const float* __restrict__ pot,
                             const int* __restrict__ seqlen,
                             const float* __restrict__ K,
                             float* __restrict__ logz_out) {
    const int lane = threadIdx.x;      // 0..31
    const int half = lane >> 4;
    const int m    = lane & 15;
    const int b0   = blockIdx.x * 16;
    const int bg   = b0 + m;
    const int L    = seqlen[bg];

    // wave-uniform max sequence length -> scalar loop bound
    int Lmax = L;
    #pragma unroll
    for (int o = 16; o >= 1; o >>= 1)
        Lmax = max(Lmax, __shfl_xor(Lmax, o, 32));
    Lmax = __builtin_amdgcn_readfirstlane(Lmax);

    // ---- constant A fragments: A[m=j_local][k=i_local] = exp(K[i][j]) ----
    v16h Af[8];
    #pragma unroll
    for (int mt = 0; mt < 4; ++mt) {
        #pragma unroll
        for (int kc = 0; kc < 2; ++kc) {
            v16h a;
            #pragma unroll
            for (int v = 0; v < 8; ++v) {
                #pragma unroll
                for (int h = 0; h < 2; ++h) {
                    int klocal = ((v < 4) ? 0 : 16) + half * 8 + (v & 3) * 2 + h;
                    int i = kc * 32 + klocal;
                    int j = mt * 16 + m;
                    a[2 * v + h] = (_Float16)__expf(K[i * Nst + j]);
                }
            }
            Af[mt * 2 + kc] = a;
        }
    }

    // ---- init: alpha0 = pot[:,0,:]; s = max, p = exp(alpha0 - s) ----
    __shared__ float sh[16 * 64];
    for (int idx = lane; idx < 16 * 64; idx += 32) {
        int bl = idx >> 6, j = idx & 63;
        sh[idx] = pot[((size_t)(b0 + bl) * Tsz) * 64 + j];
    }
    __syncthreads();
    float s = -1e30f;
    #pragma unroll 8
    for (int j = 0; j < 64; ++j) s = fmaxf(s, sh[m * 64 + j]);

    v16h Bf[2];
    #pragma unroll
    for (int kc = 0; kc < 2; ++kc) {
        v16h bb;
        #pragma unroll
        for (int v = 0; v < 8; ++v) {
            #pragma unroll
            for (int h = 0; h < 2; ++h) {
                int klocal = ((v < 4) ? 0 : 16) + half * 8 + (v & 3) * 2 + h;
                int i = kc * 32 + klocal;
                bb[2 * v + h] = (_Float16)__expf(sh[m * 64 + i] - s);
            }
        }
        Bf[kc] = bb;
    }

    const float* prow = pot + ((size_t)bg * Tsz) * 64;

    // one scan step: consume plC (pot_t), issue loads for t+1 into plN
    auto step = [&](int t, v4f (&plC)[8], v4f (&plN)[8]) {
        // pipelined loads for the NEXT step (independent -> issued early)
        int tn = min(t + 1, Tsz - 1);
        const v4f* pvn = (const v4f*)(prow + (size_t)tn * 64 + half * 8);
        #pragma unroll
        for (int mt = 0; mt < 4; ++mt) {
            plN[2 * mt]     = pvn[mt * 4];
            plN[2 * mt + 1] = pvn[mt * 4 + 1];
        }
        int tp = min(t + 4, Tsz - 1);
        __builtin_prefetch((const char*)(prow + (size_t)tp * 64) + half * 128, 0, 3);

        // q^T = eK^T * p : 8 WMMAs (4 j-tiles x 2 k-chunks)
        v8f d[4];
        #pragma unroll
        for (int mt = 0; mt < 4; ++mt) {
            v8f c = {};
            c = __builtin_amdgcn_wmma_f32_16x16x32_f16(false, Af[mt * 2 + 0],
                                                       false, Bf[0],
                                                       (short)0, c, false, false);
            c = __builtin_amdgcn_wmma_f32_16x16x32_f16(false, Af[mt * 2 + 1],
                                                       false, Bf[1],
                                                       (short)0, c, false, false);
            d[mt] = c;
        }

        // val = q * exp(pot_t)
        float val[32];
        #pragma unroll
        for (int mt = 0; mt < 4; ++mt) {
            #pragma unroll
            for (int r = 0; r < 8; ++r) {
                float pz = (r < 4) ? plC[2 * mt][r] : plC[2 * mt + 1][r - 4];
                val[mt * 8 + r] = d[mt][r] * __expf(pz);
            }
        }

        // balanced max tree (depth ~5), then partner-lane combine
        float mr[16];
        #pragma unroll
        for (int i2 = 0; i2 < 16; ++i2) mr[i2] = fmaxf(val[2 * i2], val[2 * i2 + 1]);
        #pragma unroll
        for (int w2 = 8; w2 >= 1; w2 >>= 1) {
            #pragma unroll
            for (int i2 = 0; i2 < w2; ++i2) mr[i2] = fmaxf(mr[i2], mr[i2 + w2]);
        }
        float mx = fmaxf(mr[0], __shfl_xor(mr[0], 16, 32));

        // exact power-of-two rescale: inv = 2^(126-E) -> max p in [0.5,1)
        unsigned eb  = (__float_as_uint(mx) >> 23) & 0xFFu;
        float    inv = __uint_as_float((253u - eb) << 23);
        const bool active = (t < L);
        s += active ? (float)((int)eb - 126) * 0.6931471805599453f : 0.0f;

        // per-lane D -> B repack (B0 <- tiles 0,1 ; B1 <- tiles 2,3)
        v16h nb0, nb1;
        #pragma unroll
        for (int r = 0; r < 8; ++r) {
            nb0[r]     = (_Float16)(val[r]      * inv);
            nb0[8 + r] = (_Float16)(val[8 + r]  * inv);
            nb1[r]     = (_Float16)(val[16 + r] * inv);
            nb1[8 + r] = (_Float16)(val[24 + r] * inv);
        }
        Bf[0] = active ? nb0 : Bf[0];
        Bf[1] = active ? nb1 : Bf[1];
    };

    // prologue: load pot row t=1
    v4f plA[8], plB[8];
    {
        const v4f* pv1 = (const v4f*)(prow + (size_t)1 * 64 + half * 8);
        #pragma unroll
        for (int mt = 0; mt < 4; ++mt) {
            plA[2 * mt]     = pv1[mt * 4];
            plA[2 * mt + 1] = pv1[mt * 4 + 1];
        }
    }
    // 2x-unrolled pipelined scan; overshooting Lmax by one step is identity
    for (int t = 1; t < Tsz; t += 2) {
        if (t >= Lmax) break;   // scalar, wave-uniform
        step(t,     plA, plB);
        step(t + 1, plB, plA);
    }

    // ---- logZ = s + log(sum_j p[j]) ----
    float tot = 0.0f;
    #pragma unroll
    for (int e = 0; e < 16; ++e) tot += (float)Bf[0][e] + (float)Bf[1][e];
    tot += __shfl_xor(tot, 16, 32);
    if (lane < 16) logz_out[bg] = s + __logf(tot);
}

// -------------------------------------------------------------------------
// Kernel 3: loss = mean_b( -(seq_b - logZ_b) * w_b )
// -------------------------------------------------------------------------
__global__ void crf_finalize_kernel(const float* __restrict__ seq,
                                    const float* __restrict__ logz,
                                    const float* __restrict__ w,
                                    float* __restrict__ out) {
    const int tid = threadIdx.x;  // 128 threads
    __shared__ float red[128];
    red[tid] = -(seq[tid] - logz[tid]) * w[tid];
    __syncthreads();
    #pragma unroll
    for (int s = 64; s > 0; s >>= 1) {
        if (tid < s) red[tid] += red[tid + s];
        __syncthreads();
    }
    if (tid == 0) out[0] = red[0] * (1.0f / (float)Bsz);
}

// -------------------------------------------------------------------------
extern "C" void kernel_launch(void* const* d_in, const int* in_sizes, int n_in,
                              void* d_out, int out_size, void* d_ws, size_t ws_size,
                              hipStream_t stream) {
    const float* pot  = (const float*)d_in[0];   // (128,1024,64) f32
    const int*   tags = (const int*)d_in[1];     // (128,1024) i32
    const int*   sl   = (const int*)d_in[2];     // (128,) i32
    const float* K    = (const float*)d_in[3];   // (64,64) f32
    const float* w    = (const float*)d_in[4];   // (128,) f32

    float* seq  = (float*)d_ws;        // [0..127]   sequence scores
    float* logz = seq + Bsz;           // [128..255] log-normalizers

    crf_seq_score_kernel<<<Bsz, 256, 0, stream>>>(pot, tags, sl, K, seq);
    crf_forward_scan_kernel<<<Bsz / 16, 32, 0, stream>>>(pot, sl, K, logz);
    crf_finalize_kernel<<<1, Bsz, 0, stream>>>(seq, logz, w, (float*)d_out);
}
